// VoteAggregationModule_69672959475766
// MI455X (gfx1250) — matrix operand order repaired and checked by reading, not compiled, not run on hardware
//
#include <hip/hip_runtime.h>
#include <hip/hip_bf16.h>

// ---------------------------------------------------------------------------
// VoteAggregation (PointNet++ SA module) for MI455X / gfx1250, wave32 + WMMA.
//   B=32, N=2048, C=256, S=256 proposals, ns=16 samples, radius=0.3
//   outputs: new_xyz [B,S,3], new_features [B,128,S], inds [B,S]
// ---------------------------------------------------------------------------

#define BB 32
#define NN 2048
#define CC 256
#define SS 256
#define NSAMP 16
#define R2 0.09f
#define INV_R (1.0f / 0.3f)
#define KPAD 288        // 259 channels padded to 9 chunks of K=32
#define HSTRIDE 136     // LDS row stride (f16) for 16x128 activation tiles

typedef _Float16 v16h __attribute__((ext_vector_type(16)));
typedef _Float16 v8h  __attribute__((ext_vector_type(8)));
typedef float    v8f  __attribute__((ext_vector_type(8)));

// ---------------------------------------------------------------------------
// Pack weight matrix W [Co=128, Cin] into f16 WMMA B-fragment order:
//   Bp[(((chunk*8 + tile)*32 + lane)*16) + e] = W[tile*16 + (lane&15)]
//                                               [chunk*32 + (e>>3)*16 + (lane>>4)*8 + (e&7)]
// so the GEMM loop loads one contiguous v16h (32B) per lane per wmma.
// ---------------------------------------------------------------------------
__global__ __launch_bounds__(256) void pack_weights_kernel(
    const float* __restrict__ W, _Float16* __restrict__ Bp, int Cin, int total) {
  int id = blockIdx.x * 256 + threadIdx.x;
  if (id >= total) return;
  int e     = id & 15;
  int lane  = (id >> 4) & 31;
  int tile  = (id >> 9) & 7;
  int chunk = id >> 12;
  int co = tile * 16 + (lane & 15);
  int ci = chunk * 32 + ((e >> 3) * 16) + ((lane >> 4) * 8) + (e & 7);
  float v = (ci < Cin) ? W[co * Cin + ci] : 0.0f;
  Bp[id] = (_Float16)v;
}

// Fold BN(eval) into scale/shift: y = h*sc + sh ; sc = g*rsqrt(v+eps), sh = b - m*sc
__global__ void prep_bn_kernel(
    const float* g1, const float* b1, const float* m1, const float* v1,
    const float* g2, const float* b2, const float* m2, const float* v2,
    const float* g3, const float* b3, const float* m3, const float* v3,
    float* __restrict__ sc, float* __restrict__ sh) {
  int t = threadIdx.x;            // 0..383
  if (t >= 384) return;
  int layer = t >> 7, o = t & 127;
  const float* G = (layer == 0) ? g1 : (layer == 1) ? g2 : g3;
  const float* Bt = (layer == 0) ? b1 : (layer == 1) ? b2 : b3;
  const float* M = (layer == 0) ? m1 : (layer == 1) ? m2 : m3;
  const float* V = (layer == 0) ? v1 : (layer == 1) ? v2 : v3;
  float s = G[o] * rsqrtf(V[o] + 1e-5f);
  sc[t] = s;
  sh[t] = Bt[o] - M[o] * s;
}

// ---------------------------------------------------------------------------
// features [B,C,N] f32 -> pft [B,N,KPAD] f16, channel ch = 3 + c (xyz slot at
// front filled per-group later), channels 259..287 zeroed. LDS tile transpose.
// ---------------------------------------------------------------------------
__global__ __launch_bounds__(256) void transpose_feat_kernel(
    const float* __restrict__ feat, _Float16* __restrict__ pft) {
  __shared__ _Float16 tile[32][33];
  const int b = blockIdx.z;
  const int nT = blockIdx.x * 32;   // 64 tiles over N
  const int cT = blockIdx.y * 32;   // 8 tiles over C
  const int tx = threadIdx.x, ty = threadIdx.y;
#pragma unroll
  for (int k = 0; k < 4; k++) {
    int c = cT + ty + k * 8;
    tile[ty + k * 8][tx] = (_Float16)feat[((size_t)b * CC + c) * NN + nT + tx];
  }
  __syncthreads();
#pragma unroll
  for (int k = 0; k < 4; k++) {
    int n = nT + ty + k * 8;
    pft[((size_t)b * NN + n) * KPAD + 3 + cT + tx] = tile[tx][ty + k * 8];
  }
  if (blockIdx.y == 7) {  // zero K-padding channels 259..287
#pragma unroll
    for (int k = 0; k < 4; k++) {
      int n = nT + ty + k * 8;
      if (tx < KPAD - 259)
        pft[((size_t)b * NN + n) * KPAD + 259 + tx] = (_Float16)0.0f;
    }
  }
}

// ---------------------------------------------------------------------------
// Furthest point sampling on seed_xyz, one block (256 thr) per batch.
// Seed cloud + running min-dist resident in LDS; exact first-index argmax.
// Also gathers new_xyz = xyz[b, inds] and writes the inds output.
// ---------------------------------------------------------------------------
__global__ __launch_bounds__(256) void fps_kernel(
    const float* __restrict__ seed, const float* __restrict__ xyz,
    int* __restrict__ inds_ws, float* __restrict__ nxyz_ws,
    float* __restrict__ out_xyz, float* __restrict__ out_inds) {
  const int b = blockIdx.x, t = threadIdx.x;
  __shared__ float sx[NN], sy[NN], sz[NN], dist[NN];
  __shared__ float rv[8];
  __shared__ int ri[8];
  __shared__ int sInds[SS];
  __shared__ int bestIdx;

  for (int n = t; n < NN; n += 256) {
    const float* p = seed + ((size_t)b * NN + n) * 3;
    sx[n] = p[0]; sy[n] = p[1]; sz[n] = p[2];
  }
  __syncthreads();
  const float x0 = sx[0], y0 = sy[0], z0 = sz[0];
  for (int n = t; n < NN; n += 256) {
    float dx = sx[n] - x0, dy = sy[n] - y0, dz = sz[n] - z0;
    dist[n] = dx * dx + dy * dy + dz * dz;
  }
  if (t == 0) sInds[0] = 0;
  __syncthreads();

  const int lane = t & 31, wv = t >> 5;
  for (int i = 1; i < SS; i++) {
    // local argmax (first-max-index semantics)
    float bv = -1.0f;
    int bi = 0x7fffffff;
    for (int n = t; n < NN; n += 256) {
      float v = dist[n];
      if (v > bv) { bv = v; bi = n; }
    }
    // wave32 reduction, tie-break to smaller index
    for (int off = 16; off > 0; off >>= 1) {
      float ov = __shfl_xor(bv, off, 32);
      int   oi = __shfl_xor(bi, off, 32);
      if (ov > bv || (ov == bv && oi < bi)) { bv = ov; bi = oi; }
    }
    if (lane == 0) { rv[wv] = bv; ri[wv] = bi; }
    __syncthreads();
    if (t == 0) {
      float cv = rv[0]; int ci = ri[0];
      for (int w = 1; w < 8; w++)
        if (rv[w] > cv || (rv[w] == cv && ri[w] < ci)) { cv = rv[w]; ci = ri[w]; }
      bestIdx = ci;
      sInds[i] = ci;
    }
    __syncthreads();
    const int nb = bestIdx;
    const float nx = sx[nb], ny = sy[nb], nz = sz[nb];
    for (int n = t; n < NN; n += 256) {
      float dx = sx[n] - nx, dy = sy[n] - ny, dz = sz[n] - nz;
      float d = dx * dx + dy * dy + dz * dz;
      dist[n] = fminf(dist[n], d);
    }
  }
  __syncthreads();
  // gather centers from xyz (not seed!) and emit inds
  const int ii = sInds[t];
  const int g = b * SS + t;
  inds_ws[g] = ii;
  const float* px = xyz + ((size_t)b * NN + ii) * 3;
  float X = px[0], Y = px[1], Z = px[2];
  nxyz_ws[g * 3 + 0] = X; nxyz_ws[g * 3 + 1] = Y; nxyz_ws[g * 3 + 2] = Z;
  out_xyz[g * 3 + 0] = X; out_xyz[g * 3 + 1] = Y; out_xyz[g * 3 + 2] = Z;
  out_inds[g] = (float)ii;
}

// ---------------------------------------------------------------------------
// Ball query: one wave per center; first NSAMP in-radius points in ascending
// index order (== stable argsort of the mask), padded with the first hit.
// ---------------------------------------------------------------------------
__global__ __launch_bounds__(256) void ballquery_kernel(
    const float* __restrict__ xyz, const float* __restrict__ nxyz_ws,
    int* __restrict__ idx_ws) {
  __shared__ int sidx[8][NSAMP];
  const int lane = threadIdx.x & 31, wv = threadIdx.x >> 5;
  const int g = blockIdx.x * 8 + wv;
  const int b = g >> 8;
  const float cx = nxyz_ws[g * 3 + 0], cy = nxyz_ws[g * 3 + 1], cz = nxyz_ws[g * 3 + 2];
  const float* base = xyz + (size_t)b * NN * 3;
  int cnt = 0;
  for (int s0 = 0; s0 < NN && cnt < NSAMP; s0 += 32) {
    int n = s0 + lane;
    float dx = base[n * 3 + 0] - cx;
    float dy = base[n * 3 + 1] - cy;
    float dz = base[n * 3 + 2] - cz;
    bool m = (dx * dx + dy * dy + dz * dz) < R2;
    unsigned bal = __builtin_amdgcn_ballot_w32(m);
    if (m) {
      int pos = cnt + __popc(bal & ((1u << lane) - 1u));
      if (pos < NSAMP) sidx[wv][pos] = n;
    }
    cnt += __popc(bal);
  }
  __syncthreads();
  int first = (cnt > 0) ? sidx[wv][0] : 0;
  if (lane < NSAMP) {
    int val = (lane < cnt) ? sidx[wv][lane] : first;
    idx_ws[g * NSAMP + lane] = val;
  }
}

// ---------------------------------------------------------------------------
// Fused group + 3-layer MLP + maxpool. One wave owns one group:
//   A operand = 16 samples x K channels (f16), B = packed weights,
//   D = 16x16 f32 tiles -> BN+ReLU -> LDS -> next layer's A fragments.
// WMMA: v_wmma_f32_16x16x32_f16, 8*(9+4+4) = 136 per wave.
// ---------------------------------------------------------------------------
__global__ __launch_bounds__(128) void group_mlp_kernel(
    const float* __restrict__ xyz, const _Float16* __restrict__ pft,
    const int* __restrict__ idx_ws, const float* __restrict__ nxyz_ws,
    const _Float16* __restrict__ B1p, const _Float16* __restrict__ B2p,
    const _Float16* __restrict__ B3p,
    const float* __restrict__ bnsc, const float* __restrict__ bnsh,
    float* __restrict__ out_feat) {
  __shared__ _Float16 hbuf[2][4][16 * HSTRIDE];   // ping-pong 16x128 f16/wave
  const int lane = threadIdx.x & 31, wv = threadIdx.x >> 5;
  const int g = blockIdx.x * 4 + wv;               // 8192 groups total
  const int b = g >> 8, s = g & 255;
  const int row = lane & 15;                       // sample index m / col n
  const int half = lane >> 4;
  const int kbase = half * 8;                      // A-layout K sub-offset

  // --- gather: point index, center, relative xyz ---
  const int p = idx_ws[g * NSAMP + row];
  const float cx = nxyz_ws[g * 3 + 0], cy = nxyz_ws[g * 3 + 1], cz = nxyz_ws[g * 3 + 2];
  const float* pxyz = xyz + ((size_t)b * NN + p) * 3;
  float gx0 = (pxyz[0] - cx) * INV_R;
  float gx1 = (pxyz[1] - cy) * INV_R;
  float gx2 = (pxyz[2] - cz) * INV_R;
  const _Float16* fp = pft + ((size_t)b * NN + p) * KPAD;

  // --- build layer-1 A fragments: 9 chunks of 16x32 f16 ---
  v16h A[9];
  {
    union { v16h v; _Float16 h[16]; } u;
#pragma unroll
    for (int e = 0; e < 16; e++) {
      int ch = (e < 8) ? (kbase + e) : (16 + kbase + (e - 8));
      float val;
      if (e < 3) {
        val = (half == 0) ? ((e == 0) ? gx0 : (e == 1) ? gx1 : gx2)
                          : (float)fp[ch];
      } else {
        val = (float)fp[ch];
      }
      u.h[e] = (_Float16)val;
    }
    A[0] = u.v;
  }
#pragma unroll
  for (int c = 1; c < 9; c++) {
    union { v16h v; v8h h[2]; } u;
    u.h[0] = *(const v8h*)(fp + c * 32 + kbase);
    u.h[1] = *(const v8h*)(fp + c * 32 + 16 + kbase);
    A[c] = u.v;
  }

  // --- layer 1: [16 x 288] x [288 x 128] ---
  _Float16* h0 = hbuf[0][wv];
#pragma unroll
  for (int ti = 0; ti < 8; ti++) {
    v8f acc = {};
#pragma unroll
    for (int c = 0; c < 9; c++) {
      v16h bf = *(const v16h*)(B1p + (((c * 8 + ti) * 32 + lane) << 4));
      acc = __builtin_amdgcn_wmma_f32_16x16x32_f16(
          false, A[c], false, bf, (short)0, acc, false, false);
    }
    const int co = ti * 16 + row;
    const float sc = bnsc[co], sh = bnsh[co];
#pragma unroll
    for (int r = 0; r < 8; r++) {
      int m = (half << 3) + r;
      float y = fmaxf(acc[r] * sc + sh, 0.0f);
      h0[m * HSTRIDE + co] = (_Float16)y;
    }
  }
  __syncthreads();

  // --- A fragments for layer 2 from LDS (layout conversion C/D -> A) ---
  v16h A2[4];
#pragma unroll
  for (int c = 0; c < 4; c++) {
    union { v16h v; v8h h[2]; } u;
    u.h[0] = *(const v8h*)(h0 + row * HSTRIDE + c * 32 + kbase);
    u.h[1] = *(const v8h*)(h0 + row * HSTRIDE + c * 32 + 16 + kbase);
    A2[c] = u.v;
  }

  // --- layer 2 ---
  _Float16* h1 = hbuf[1][wv];
#pragma unroll
  for (int ti = 0; ti < 8; ti++) {
    v8f acc = {};
#pragma unroll
    for (int c = 0; c < 4; c++) {
      v16h bf = *(const v16h*)(B2p + (((c * 8 + ti) * 32 + lane) << 4));
      acc = __builtin_amdgcn_wmma_f32_16x16x32_f16(
          false, A2[c], false, bf, (short)0, acc, false, false);
    }
    const int co = ti * 16 + row;
    const float sc = bnsc[128 + co], sh = bnsh[128 + co];
#pragma unroll
    for (int r = 0; r < 8; r++) {
      int m = (half << 3) + r;
      float y = fmaxf(acc[r] * sc + sh, 0.0f);
      h1[m * HSTRIDE + co] = (_Float16)y;
    }
  }
  __syncthreads();

  // --- A fragments for layer 3 ---
  v16h A3[4];
#pragma unroll
  for (int c = 0; c < 4; c++) {
    union { v16h v; v8h h[2]; } u;
    u.h[0] = *(const v8h*)(h1 + row * HSTRIDE + c * 32 + kbase);
    u.h[1] = *(const v8h*)(h1 + row * HSTRIDE + c * 32 + 16 + kbase);
    A3[c] = u.v;
  }

  // --- layer 3 + max over the 16 samples ---
#pragma unroll
  for (int ti = 0; ti < 8; ti++) {
    v8f acc = {};
#pragma unroll
    for (int c = 0; c < 4; c++) {
      v16h bf = *(const v16h*)(B3p + (((c * 8 + ti) * 32 + lane) << 4));
      acc = __builtin_amdgcn_wmma_f32_16x16x32_f16(
          false, A3[c], false, bf, (short)0, acc, false, false);
    }
    const int co = ti * 16 + row;
    const float sc = bnsc[256 + co], sh = bnsh[256 + co];
    float mx = 0.0f;  // post-ReLU values are >= 0
#pragma unroll
    for (int r = 0; r < 8; r++) {
      float y = fmaxf(acc[r] * sc + sh, 0.0f);
      mx = fmaxf(mx, y);
    }
    float other = __shfl_xor(mx, 16, 32);   // combine the two 8-row halves
    mx = fmaxf(mx, other);
    if (lane < 16)
      out_feat[((size_t)b * 128 + co) * SS + s] = mx;
  }
}

// ---------------------------------------------------------------------------
extern "C" void kernel_launch(void* const* d_in, const int* in_sizes, int n_in,
                              void* d_out, int out_size, void* d_ws, size_t ws_size,
                              hipStream_t stream) {
  (void)in_sizes; (void)n_in; (void)out_size; (void)ws_size;
  const float* xyz  = (const float*)d_in[0];
  const float* feat = (const float*)d_in[1];
  const float* seed = (const float*)d_in[2];
  const float* W1 = (const float*)d_in[3];
  const float* g1 = (const float*)d_in[4];
  const float* b1 = (const float*)d_in[5];
  const float* m1 = (const float*)d_in[6];
  const float* v1 = (const float*)d_in[7];
  const float* W2 = (const float*)d_in[8];
  const float* g2 = (const float*)d_in[9];
  const float* b2 = (const float*)d_in[10];
  const float* m2 = (const float*)d_in[11];
  const float* v2 = (const float*)d_in[12];
  const float* W3 = (const float*)d_in[13];
  const float* g3 = (const float*)d_in[14];
  const float* b3 = (const float*)d_in[15];
  const float* m3 = (const float*)d_in[16];
  const float* v3 = (const float*)d_in[17];

  float* out = (float*)d_out;
  float* out_xyz  = out;                       // [B,S,3]   24576
  float* out_feat = out + BB * SS * 3;         // [B,128,S] 1048576
  float* out_inds = out_feat + BB * 128 * SS;  // [B,S]     8192 (as float)

  // workspace carve-out
  char* ws = (char*)d_ws;
  size_t off = 0;
  int* inds_ws = (int*)(ws + off);        off += (size_t)BB * SS * 4;
  int* idx_ws  = (int*)(ws + off);        off += (size_t)BB * SS * NSAMP * 4;
  float* nxyz_ws = (float*)(ws + off);    off += (size_t)BB * SS * 3 * 4;
  _Float16* B1p = (_Float16*)(ws + off);  off += (size_t)9 * 8 * 32 * 16 * 2;
  _Float16* B2p = (_Float16*)(ws + off);  off += (size_t)4 * 8 * 32 * 16 * 2;
  _Float16* B3p = (_Float16*)(ws + off);  off += (size_t)4 * 8 * 32 * 16 * 2;
  float* bnsc = (float*)(ws + off);       off += 384 * 4;
  float* bnsh = (float*)(ws + off);       off += 384 * 4;
  off = (off + 255) & ~(size_t)255;
  _Float16* pft = (_Float16*)(ws + off);  // [B,N,KPAD] f16 ~ 36 MB

  // 1) weight / BN prep (tiny, L2 resident afterwards)
  pack_weights_kernel<<<144, 256, 0, stream>>>(W1, B1p, CC + 3, 9 * 8 * 32 * 16);
  pack_weights_kernel<<<64, 256, 0, stream>>>(W2, B2p, 128, 4 * 8 * 32 * 16);
  pack_weights_kernel<<<64, 256, 0, stream>>>(W3, B3p, 128, 4 * 8 * 32 * 16);
  prep_bn_kernel<<<1, 384, 0, stream>>>(g1, b1, m1, v1, g2, b2, m2, v2,
                                        g3, b3, m3, v3, bnsc, bnsh);

  // 2) features [B,C,N] f32 -> padded [B,N,288] f16 (coalesced LDS transpose)
  transpose_feat_kernel<<<dim3(NN / 32, CC / 32, BB), dim3(32, 8), 0, stream>>>(feat, pft);

  // 3) FPS on seeds (LDS-resident, one block/batch) + center gather + inds out
  fps_kernel<<<BB, 256, 0, stream>>>(seed, xyz, inds_ws, nxyz_ws, out_xyz, out_inds);

  // 4) ball query: one wave per center
  ballquery_kernel<<<(BB * SS) / 8, 256, 0, stream>>>(xyz, nxyz_ws, idx_ws);

  // 5) fused group + MLP (WMMA) + maxpool: one wave per group, 4 waves/block
  group_mlp_kernel<<<(BB * SS) / 4, 128, 0, stream>>>(
      xyz, pft, idx_ws, nxyz_ws, B1p, B2p, B3p, bnsc, bnsh, out_feat);
}
